// MultiImageVisionTransformer_89086211653901
// MI455X (gfx1250) — matrix operand order repaired and checked by reading, not compiled, not run on hardware
//
#include <hip/hip_runtime.h>
#include <hip/hip_bf16.h>
#include <cstdint>

typedef unsigned short u16;
typedef __attribute__((ext_vector_type(16))) __bf16 v16bf;
typedef __attribute__((ext_vector_type(8)))  float  v8f;

#define EMB   768
#define HID   3072
#define NHEAD 12
#define HD    64
#define NTOK  784
#define NPP   196
#define NB    8
#define QSZ   (NB*NHEAD*NTOK*HD)   /* 4,816,896 elements per Q/K/V buffer */

struct RowMap { int n; int bs; int off; };

__device__ __forceinline__ int prow(const RowMap& rm, int m) {
  return (m / rm.n) * rm.bs + (m % rm.n) + rm.off;
}

__device__ __forceinline__ u16 f2bf(float x) {
  union { float f; unsigned int u; } c; c.f = x;
  unsigned int u = c.u;
  u += 0x7fffu + ((u >> 16) & 1u);
  return (u16)(u >> 16);
}

__device__ __forceinline__ float gelu_f(float x) {
  return 0.5f * x * (1.0f + erff(x * 0.70710678118f));
}

union FragAB { uint4 u[2]; v16bf v; };

// ---------------------------------------------------------------------------
// Generic bf16 WMMA GEMM:  C[M,N] = A[M,K] * W[N,K]^T (+ epilogue)
// Block = 256 threads (8 wave32), tile 256(M) x 64(N), K stepped by 32.
// Each wave computes a 32x64 strip: 2 A-frags x 4 B-frags = 8 WMMAs/K-step,
// software-pipelined LDS staging (next tile loads overlap current WMMAs).
// MODE 1: fp32 residual in-place (Cf += bias + acc)
// MODE 2: gelu(acc+bias) -> bf16 Cb
// MODE 3: qkv scatter -> Q[b,h,t,d], K[b,h,t,d], Vt[b,h,d,t] (bf16)
// MODE 4: patch-embed: acc + bias + pos -> fp32 Cf
// ---------------------------------------------------------------------------
template<int MODE>
__global__ void __launch_bounds__(256)
gemm_wmma_k(const u16* __restrict__ A, const u16* __restrict__ Bw,
            const float* __restrict__ bias,
            float* Cf, u16* Cb,
            const float* __restrict__ aux0, const float* __restrict__ aux1,
            int M, int N, int K, int ldA, int ldC,
            RowMap amap, RowMap cmap, int auxi)
{
  __shared__ __align__(16) u16 As[256 * 32];   // 16 KB
  __shared__ __align__(16) u16 Bs[64 * 32];    //  4 KB

  const int tid  = threadIdx.x;
  const int m0   = blockIdx.x * 256;
  const int n0   = blockIdx.y * 64;
  const int w    = tid >> 5;
  const int lane = tid & 31;
  const int lr   = lane & 15;
  const int hs   = lane >> 4;

  v8f acc[2][4];
  for (int u = 0; u < 2; ++u)
    for (int f = 0; f < 4; ++f)
      for (int r = 0; r < 8; ++r) acc[u][f][r] = 0.0f;

  // staging-load coordinates (fixed per thread): one 32-elem A row + 8 B elems
  int arow = m0 + tid;
  if (arow >= M) arow = M - 1;
  const long aoff = (long)prow(amap, arow) * ldA;
  const long boff = (long)(n0 + (tid >> 2)) * K;
  const int  bkc  = (tid & 3) * 8;

  // prologue: load k0 = 0 tile into registers
  uint4 a_reg[4];
  uint4 b_reg;
  {
    const uint4* asrc = (const uint4*)(A + aoff);
    a_reg[0] = asrc[0]; a_reg[1] = asrc[1]; a_reg[2] = asrc[2]; a_reg[3] = asrc[3];
    b_reg = *(const uint4*)(Bw + boff + bkc);
  }

  for (int k0 = 0; k0 < K; k0 += 32) {
    // commit staged registers to LDS
    {
      uint4* dst = (uint4*)(As + tid * 32);
      dst[0] = a_reg[0]; dst[1] = a_reg[1]; dst[2] = a_reg[2]; dst[3] = a_reg[3];
      *(uint4*)(Bs + (tid >> 2) * 32 + bkc) = b_reg;
    }
    __syncthreads();

    // start next tile's global loads (overlap with WMMA below)
    {
      const int kn = (k0 + 32 < K) ? k0 + 32 : k0;
      const uint4* asrc = (const uint4*)(A + aoff + kn);
      a_reg[0] = asrc[0]; a_reg[1] = asrc[1]; a_reg[2] = asrc[2]; a_reg[3] = asrc[3];
      b_reg = *(const uint4*)(Bw + boff + kn + bkc);
      if (k0 + 64 < K) {
        __builtin_prefetch(A + aoff + k0 + 64, 0, 0);
        __builtin_prefetch(Bw + boff + k0 + 64, 0, 0);
      }
    }

    // fragments from LDS: 2 A-frags (rows w*32 .. w*32+31), 4 B-frags
    FragAB fa[2];
    for (int u = 0; u < 2; ++u) {
      const u16* ap = As + (w * 32 + u * 16 + lr) * 32 + hs * 8;
      fa[u].u[0] = *(const uint4*)ap;
      fa[u].u[1] = *(const uint4*)(ap + 16);
    }
    for (int f = 0; f < 4; ++f) {
      FragAB fb;
      const u16* bp = Bs + (f * 16 + lr) * 32 + hs * 8;
      fb.u[0] = *(const uint4*)bp;
      fb.u[1] = *(const uint4*)(bp + 16);
      acc[0][f] = __builtin_amdgcn_wmma_f32_16x16x32_bf16(
                      false, fa[0].v, false, fb.v, (short)0, acc[0][f], false, false);
      acc[1][f] = __builtin_amdgcn_wmma_f32_16x16x32_bf16(
                      false, fa[1].v, false, fb.v, (short)0, acc[1][f], false, false);
    }
    __syncthreads();
  }

  // epilogue
  for (int u = 0; u < 2; ++u) {
    const int gmb = m0 + w * 32 + u * 16 + hs * 8;
    for (int f = 0; f < 4; ++f) {
      const int gn = n0 + f * 16 + lr;
      const float bv0 = bias ? bias[gn] : 0.0f;
      for (int r = 0; r < 8; ++r) {
        const int gm = gmb + r;
        if (gm >= M) continue;
        float v = acc[u][f][r] + bv0;
        if (MODE == 1) {
          long idx = (long)prow(cmap, gm) * ldC + gn;
          Cf[idx] = Cf[idx] + v;
        } else if (MODE == 2) {
          long idx = (long)prow(cmap, gm) * ldC + gn;
          Cb[idx] = f2bf(gelu_f(v));
        } else if (MODE == 3) {
          const int b = gm / cmap.n, t = gm % cmap.n;
          const int j = gn / EMB, rem = gn % EMB;
          const int h = rem / HD, dd = rem % HD;
          const long bh = (long)(b * NHEAD + h);
          const u16 bv = f2bf(v);
          if (j == 0)       Cb[(bh * NTOK + t) * HD + dd] = bv;
          else if (j == 1)  Cb[(long)QSZ + (bh * NTOK + t) * HD + dd] = bv;
          else              Cb[2L * QSZ + (bh * HD + dd) * NTOK + t] = bv;
        } else if (MODE == 4) {
          const int t = gm % cmap.n;
          const float pos = (gn < EMB / 2)
              ? aux0[(long)t * (EMB / 2) + gn]
              : aux1[(long)auxi * (EMB / 2) + (gn - EMB / 2)];
          long idx = (long)prow(cmap, gm) * ldC + gn;
          Cf[idx] = v + pos;
        }
      }
    }
  }
}

// ---------------------------------------------------------------------------
// Fused flash attention: one wave per (b, h, 16-query tile).
// Streams keys in tiles of 32, online softmax, P transposed through LDS.
// Q/K layout: [b,h,t,d] bf16 ; V layout (pre-transposed): [b,h,d,t] bf16.
// ---------------------------------------------------------------------------
__global__ void __launch_bounds__(32)
attn_wmma_k(const u16* __restrict__ Q, const u16* __restrict__ Kh,
            const u16* __restrict__ Vt, u16* __restrict__ Ob, int ncur)
{
  __shared__ __align__(16) u16 Plds[16 * 32];

  const int lane = threadIdx.x & 31;
  const int lr = lane & 15, hs = lane >> 4;
  const int q0 = blockIdx.x * 16;
  const int h  = blockIdx.y;
  const int b  = blockIdx.z;
  const long bh = (long)(b * NHEAD + h);
  const u16* qb = Q  + bh * NTOK * HD;
  const u16* kb = Kh + bh * NTOK * HD;
  const u16* vb = Vt + bh * HD * NTOK;

  // Q A-fragments (d=0..31, d=32..63), loaded once
  int tq = q0 + lr; if (tq >= ncur) tq = ncur - 1;
  FragAB qa[2];
  for (int s = 0; s < 2; ++s) {
    const u16* p = qb + (long)tq * HD + s * 32 + hs * 8;
    qa[s].u[0] = *(const uint4*)p;
    qa[s].u[1] = *(const uint4*)(p + 16);
  }

  v8f O[4];
  for (int f = 0; f < 4; ++f)
    for (int r = 0; r < 8; ++r) O[f][r] = 0.0f;
  float m_st[8], l_st[8];
  for (int r = 0; r < 8; ++r) { m_st[r] = -1e30f; l_st[r] = 0.0f; }

  for (int j0 = 0; j0 < ncur; j0 += 32) {
    // ---- scores: S[16 x 32] = Q (16x64) . K^T ----
    v8f sc[2];
    for (int f = 0; f < 2; ++f)
      for (int r = 0; r < 8; ++r) sc[f][r] = 0.0f;
    for (int f = 0; f < 2; ++f) {
      int kt = j0 + f * 16 + lr; if (kt >= ncur) kt = ncur - 1;
      for (int s = 0; s < 2; ++s) {
        FragAB fb;
        const u16* p = kb + (long)kt * HD + s * 32 + hs * 8;
        fb.u[0] = *(const uint4*)p;
        fb.u[1] = *(const uint4*)(p + 16);
        sc[f] = __builtin_amdgcn_wmma_f32_16x16x32_bf16(
                    false, qa[s].v, false, fb.v, (short)0, sc[f], false, false);
      }
    }
    // scale + key mask
    const bool ok0 = (j0 + lr) < ncur;
    const bool ok1 = (j0 + 16 + lr) < ncur;
    for (int r = 0; r < 8; ++r) {
      sc[0][r] = ok0 ? sc[0][r] * 0.125f : -1e30f;
      sc[1][r] = ok1 ? sc[1][r] * 0.125f : -1e30f;
    }
    // ---- online softmax (row stats via half-wave shfl reductions) ----
    float mt[8];
    for (int r = 0; r < 8; ++r) mt[r] = fmaxf(sc[0][r], sc[1][r]);
    for (int s = 1; s < 16; s <<= 1)
      for (int r = 0; r < 8; ++r) mt[r] = fmaxf(mt[r], __shfl_xor(mt[r], s));
    float alpha[8];
    for (int r = 0; r < 8; ++r) {
      float mn = fmaxf(m_st[r], mt[r]);
      alpha[r] = __expf(m_st[r] - mn);
      m_st[r]  = mn;
    }
    for (int r = 0; r < 8; ++r) {
      sc[0][r] = __expf(sc[0][r] - m_st[r]);
      sc[1][r] = __expf(sc[1][r] - m_st[r]);
    }
    float st[8];
    for (int r = 0; r < 8; ++r) st[r] = sc[0][r] + sc[1][r];
    for (int s = 1; s < 16; s <<= 1)
      for (int r = 0; r < 8; ++r) st[r] += __shfl_xor(st[r], s);
    for (int r = 0; r < 8; ++r) l_st[r] = l_st[r] * alpha[r] + st[r];
    for (int f = 0; f < 4; ++f)
      for (int r = 0; r < 8; ++r) O[f][r] *= alpha[r];

    // ---- P (C-layout) -> LDS -> A-fragment layout ----
    __syncthreads();
    for (int r = 0; r < 8; ++r) {
      const int row = hs * 8 + r;
      Plds[row * 32 + lr]      = f2bf(sc[0][r]);
      Plds[row * 32 + 16 + lr] = f2bf(sc[1][r]);
    }
    __syncthreads();
    FragAB pa;
    const u16* pp = Plds + lr * 32 + hs * 8;
    pa.u[0] = *(const uint4*)pp;
    pa.u[1] = *(const uint4*)(pp + 16);

    // ---- O += P (16x32) . V (32x64); Vt is d-major so B-frags are contiguous
    for (int f = 0; f < 4; ++f) {
      FragAB vf;
      const u16* p = vb + (long)(f * 16 + lr) * NTOK + j0 + hs * 8;
      vf.u[0] = *(const uint4*)p;
      vf.u[1] = *(const uint4*)(p + 16);
      O[f] = __builtin_amdgcn_wmma_f32_16x16x32_bf16(
                 false, pa.v, false, vf.v, (short)0, O[f], false, false);
    }
  }

  for (int f = 0; f < 4; ++f)
    for (int r = 0; r < 8; ++r) {
      const int t = q0 + hs * 8 + r;
      if (t < ncur) {
        const float v = O[f][r] / (l_st[r] + 1e-20f);
        Ob[((long)(b * NTOK + t)) * EMB + h * HD + f * 16 + lr] = f2bf(v);
      }
    }
}

// ---------------------------------------------------------------------------
// LayerNorm over EMB=768; one row per block. Writes bf16 (ob) and/or fp32 (of,
// may alias input for in-place).
// ---------------------------------------------------------------------------
__global__ void __launch_bounds__(256)
ln_k(const float* __restrict__ X, const float* __restrict__ g,
     const float* __restrict__ be, u16* ob, float* of, int ncur)
{
  __shared__ float red[256];
  const int m = blockIdx.x;
  const long row = (long)((m / ncur) * NTOK + (m % ncur)) * EMB;
  const int tid = threadIdx.x;

  float s = 0.0f;
  for (int e = tid; e < EMB; e += 256) s += X[row + e];
  red[tid] = s; __syncthreads();
  for (int st = 128; st; st >>= 1) { if (tid < st) red[tid] += red[tid + st]; __syncthreads(); }
  const float mu = red[0] * (1.0f / EMB);
  __syncthreads();

  float v = 0.0f;
  for (int e = tid; e < EMB; e += 256) { float d = X[row + e] - mu; v += d * d; }
  red[tid] = v; __syncthreads();
  for (int st = 128; st; st >>= 1) { if (tid < st) red[tid] += red[tid + st]; __syncthreads(); }
  const float rs = rsqrtf(red[0] * (1.0f / EMB) + 1e-5f);

  for (int e = tid; e < EMB; e += 256) {
    const float y = (X[row + e] - mu) * rs * g[e] + be[e];
    if (ob) ob[row + e] = f2bf(y);
    if (of) of[row + e] = y;
  }
}

// ---------------------------------------------------------------------------
__global__ void cvt_k(const float* __restrict__ s, u16* __restrict__ d, long n) {
  for (long i = (long)blockIdx.x * 256 + threadIdx.x; i < n; i += (long)gridDim.x * 256)
    d[i] = f2bf(s[i]);
}

__global__ void im2col_k(const float* __restrict__ img, u16* __restrict__ P, int n) {
  const int i = blockIdx.x * 256 + threadIdx.x;
  if (i >= n) return;
  const int m = i / EMB, k = i % EMB;
  const int b = m / NPP, p = m % NPP;
  const int gr = p / 14, gc = p % 14;
  const int c = k >> 8, rr = (k >> 4) & 15, cc = k & 15;
  const float v = img[(((long)b * 3 + c) * 224 + gr * 16 + rr) * 224 + gc * 16 + cc];
  P[i] = f2bf(v);
}

__global__ void mean_k(const float* __restrict__ X, float* __restrict__ xm) {
  const int b = blockIdx.x;
  const int e = blockIdx.y * 256 + threadIdx.x;
  float s = 0.0f;
  for (int t = 0; t < NTOK; ++t) s += X[((long)(b * NTOK + t)) * EMB + e];
  xm[b * EMB + e] = s * (1.0f / NTOK);
}

__global__ void head1_k(const float* __restrict__ xm, const float* __restrict__ w,
                        const float* __restrict__ bi, float* __restrict__ h) {
  const int b = blockIdx.x;
  const int e = blockIdx.y * 256 + threadIdx.x;
  float s = bi[e];
  const float* xr = xm + b * EMB;
  const float* wr = w + (long)e * EMB;
  for (int k = 0; k < EMB; ++k) s += xr[k] * wr[k];
  h[b * EMB + e] = gelu_f(s);
}

__global__ void head2_k(const float* __restrict__ h, const float* __restrict__ w,
                        const float* __restrict__ bi, float* __restrict__ pat) {
  const int b = blockIdx.x;
  const int p = threadIdx.x;   // 256 = PATCH*PATCH
  float s = bi[p];
  const float* hr = h + b * EMB;
  const float* wr = w + (long)p * EMB;
  for (int k = 0; k < EMB; ++k) s += hr[k] * wr[k];
  pat[b * 256 + p] = s;
}

__global__ void bcast_k(const float* __restrict__ pat, float* __restrict__ out, int n) {
  const int i = blockIdx.x * 256 + threadIdx.x;
  if (i >= n) return;
  const int b = i / (224 * 224);
  const int rc = i % (224 * 224);
  const int row = rc / 224, col = rc % 224;
  out[i] = pat[b * 256 + (row & 15) * 16 + (col & 15)];
}

// ---------------------------------------------------------------------------
extern "C" void kernel_launch(void* const* d_in, const int* in_sizes, int n_in,
                              void* d_out, int out_size, void* d_ws, size_t ws_size,
                              hipStream_t stream)
{
  (void)in_sizes; (void)n_in; (void)out_size; (void)ws_size;
  const float* img[4]    = {(const float*)d_in[0], (const float*)d_in[1],
                            (const float*)d_in[2], (const float*)d_in[3]};
  const float* pe_w      = (const float*)d_in[4];
  const float* pe_b      = (const float*)d_in[5];
  const float* spatial   = (const float*)d_in[6];
  const float* imagep    = (const float*)d_in[7];
  const float* ln_attn_g = (const float*)d_in[8];
  const float* ln_attn_b = (const float*)d_in[9];
  const float* qkv_w     = (const float*)d_in[10];
  const float* qkv_b     = (const float*)d_in[11];
  const float* proj_w    = (const float*)d_in[12];
  const float* proj_b    = (const float*)d_in[13];
  const float* ln1_g     = (const float*)d_in[14];
  const float* ln1_b     = (const float*)d_in[15];
  const float* fc1_w     = (const float*)d_in[16];
  const float* fc1_b     = (const float*)d_in[17];
  const float* fc2_w     = (const float*)d_in[18];
  const float* fc2_b     = (const float*)d_in[19];
  const float* ln2_g     = (const float*)d_in[20];
  const float* ln2_b     = (const float*)d_in[21];
  const float* final_g   = (const float*)d_in[22];
  const float* final_b   = (const float*)d_in[23];
  const float* head_w1   = (const float*)d_in[24];
  const float* head_b1   = (const float*)d_in[25];
  const float* head_w2   = (const float*)d_in[26];
  const float* head_b2   = (const float*)d_in[27];
  float* out = (float*)d_out;

  char* ws = (char*)d_ws;
  size_t off = 0;
  auto alloc = [&](size_t nbytes) {
    off = (off + 255) & ~(size_t)255;
    size_t r = off; off += nbytes; return r;
  };
  u16*   pewB  = (u16*)(ws + alloc((size_t)4 * EMB * EMB * 2));
  u16*   qkvB  = (u16*)(ws + alloc((size_t)3 * 3 * EMB * EMB * 2));
  u16*   projB = (u16*)(ws + alloc((size_t)3 * EMB * EMB * 2));
  u16*   fc1B  = (u16*)(ws + alloc((size_t)3 * HID * EMB * 2));
  u16*   fc2B  = (u16*)(ws + alloc((size_t)3 * EMB * HID * 2));
  float* X     = (float*)(ws + alloc((size_t)NB * NTOK * EMB * 4));
  u16*   Hb    = (u16*)(ws + alloc((size_t)NB * NTOK * EMB * 2));
  u16*   Qh    = (u16*)(ws + alloc((size_t)3 * QSZ * 2 + 256));  // +pad for V edge reads
  u16*   Ob    = (u16*)(ws + alloc((size_t)NB * NTOK * EMB * 2));
  u16*   G     = (u16*)(ws + alloc((size_t)NB * NTOK * HID * 2));
  u16*   Pim   = (u16*)(ws + alloc((size_t)NB * NPP * EMB * 2 + 256));
  float* xm    = (float*)(ws + alloc((size_t)NB * EMB * 4));
  float* h1    = (float*)(ws + alloc((size_t)NB * EMB * 4));
  float* pat   = (float*)(ws + alloc((size_t)NB * 256 * 4));

  auto cvt = [&](const float* s, u16* d, long n) {
    long blocks = (n + 255) / 256; if (blocks > 8192) blocks = 8192;
    cvt_k<<<(int)blocks, 256, 0, stream>>>(s, d, n);
  };
  cvt(pe_w,  pewB,  (long)4 * EMB * EMB);
  cvt(qkv_w, qkvB,  (long)3 * 3 * EMB * EMB);
  cvt(proj_w,projB, (long)3 * EMB * EMB);
  cvt(fc1_w, fc1B,  (long)3 * HID * EMB);
  cvt(fc2_w, fc2B,  (long)3 * EMB * HID);

  auto embed = [&](int idx) {
    const int n = NB * NPP * EMB;
    im2col_k<<<(n + 255) / 256, 256, 0, stream>>>(img[idx], Pim, n);
    dim3 g((NB * NPP + 255) / 256, EMB / 64);
    gemm_wmma_k<4><<<g, 256, 0, stream>>>(
        Pim, pewB + (size_t)idx * EMB * EMB, pe_b + idx * EMB,
        X, nullptr, spatial, imagep,
        NB * NPP, EMB, EMB, EMB, EMB,
        RowMap{NPP, NPP, 0}, RowMap{NPP, NTOK, idx * NPP}, idx);
  };

  auto block = [&](int i, int ncur) {
    const int M = NB * ncur;
    const RowMap rm{ncur, NTOK, 0};
    ln_k<<<M, 256, 0, stream>>>(X, ln_attn_g + i * EMB, ln_attn_b + i * EMB, Hb, nullptr, ncur);
    gemm_wmma_k<3><<<dim3((M + 255) / 256, (3 * EMB) / 64), 256, 0, stream>>>(
        Hb, qkvB + (size_t)i * 3 * EMB * EMB, qkv_b + i * 3 * EMB,
        nullptr, Qh, nullptr, nullptr, M, 3 * EMB, EMB, EMB, 0, rm, rm, 0);
    attn_wmma_k<<<dim3((ncur + 15) / 16, NHEAD, NB), 32, 0, stream>>>(
        Qh, Qh + QSZ, Qh + 2L * QSZ, Ob, ncur);
    gemm_wmma_k<1><<<dim3((M + 255) / 256, EMB / 64), 256, 0, stream>>>(
        Ob, projB + (size_t)i * EMB * EMB, proj_b + i * EMB,
        X, nullptr, nullptr, nullptr, M, EMB, EMB, EMB, EMB, rm, rm, 0);
    ln_k<<<M, 256, 0, stream>>>(X, ln1_g + i * EMB, ln1_b + i * EMB, Hb, nullptr, ncur);
    gemm_wmma_k<2><<<dim3((M + 255) / 256, HID / 64), 256, 0, stream>>>(
        Hb, fc1B + (size_t)i * HID * EMB, fc1_b + i * HID,
        nullptr, G, nullptr, nullptr, M, HID, EMB, EMB, HID, rm, rm, 0);
    gemm_wmma_k<1><<<dim3((M + 255) / 256, EMB / 64), 256, 0, stream>>>(
        G, fc2B + (size_t)i * EMB * HID, fc2_b + i * EMB,
        X, nullptr, nullptr, nullptr, M, EMB, HID, HID, EMB, rm, rm, 0);
    ln_k<<<M, 256, 0, stream>>>(X, ln2_g + i * EMB, ln2_b + i * EMB, nullptr, X, ncur);
  };

  embed(0); embed(1);
  block(0, 2 * NPP);
  embed(2); block(1, 3 * NPP);
  embed(3); block(2, 4 * NPP);

  ln_k<<<NB * NTOK, 256, 0, stream>>>(X, final_g, final_b, nullptr, X, NTOK);
  mean_k<<<dim3(NB, EMB / 256), 256, 0, stream>>>(X, xm);
  head1_k<<<dim3(NB, EMB / 256), 256, 0, stream>>>(xm, head_w1, head_b1, h1);
  head2_k<<<NB, 256, 0, stream>>>(h1, head_w2, head_b2, pat);
  const int no = NB * 224 * 224;
  bcast_k<<<(no + 255) / 256, 256, 0, stream>>>(pat, out, no);
}
